// TransformerBlock_79302276153669
// MI455X (gfx1250) — matrix operand order repaired and checked by reading, not compiled, not run on hardware
//
#include <hip/hip_runtime.h>
#include <hip/hip_bf16.h>
#include <math.h>
#include <stdint.h>

typedef __attribute__((ext_vector_type(16))) _Float16 v16h;
typedef __attribute__((ext_vector_type(8)))  _Float16 v8h;
typedef __attribute__((ext_vector_type(8)))  float    v8f;

#define BDIM 2
#define TDIM 2048
#define DDIM 1024
#define HDIM 16
#define HD   64
#define DFF  4096
#define MROWS (BDIM*TDIM)   // 4096

// ---------------------------------------------------------------- helpers ---
__device__ inline v8f wmma_f16f32(v16h a, v16h b, v8f c) {
  return __builtin_amdgcn_wmma_f32_16x16x32_f16(
      false, a, false, b, (short)0, c, false, false);
}

__device__ inline v8f vzero8() {
  v8f z;
#pragma unroll
  for (int i = 0; i < 8; ++i) z[i] = 0.0f;
  return z;
}

// Async global->LDS copy (CDNA5 GLOBAL_LOAD_ASYNC_TO_LDS_B128, ASYNCcnt).
// LDS operand = low 32 bits of the flat shared-memory pointer (ISA: LDS
// aperture address truncates to addr[31:0]).
__device__ inline void async_ld_b128(const _Float16* lds_dst, const _Float16* g) {
  uint32_t lds = (uint32_t)(size_t)lds_dst;
  asm volatile("global_load_async_to_lds_b128 %0, %1, off"
               :: "v"(lds), "v"(g) : "memory");
}
#define ASYNC_WAIT_ALL()  asm volatile("s_wait_asynccnt 0x0" ::: "memory")
#define ASYNC_WAIT_KEEP4() asm volatile("s_wait_asynccnt 0x4" ::: "memory")

// A-fragment (16x32 f16). Lane L: m = L&15.
// lanes 0-15:  elems 0..7 = K0..7,  elems 8..15 = K16..23
// lanes 16-31: elems 0..7 = K8..15, elems 8..15 = K24..31
__device__ inline v16h load_frag_a(const _Float16* base, int stride, int lane) {
  const int m   = lane & 15;
  const int klo = (lane < 16) ? 0 : 8;
  const _Float16* p = base + (size_t)m * stride + klo;
  v8h lo = *(const v8h*)(p);
  v8h hi = *(const v8h*)(p + 16);
  v16h r;
#pragma unroll
  for (int i = 0; i < 8; ++i) { r[i] = lo[i]; r[i + 8] = hi[i]; }
  return r;
}

// B-fragment (32x16 f16). Lane L: n = L&15.
// lanes 0-15: elems 0..7 = K0..7, 8..15 = K16..23 ; lanes 16-31: elems 0..7 = K8..15, 8..15 = K24..31
__device__ inline v16h load_frag_b(const _Float16* base, int stride, int lane) {
  const int n    = lane & 15;
  const int koff = (lane < 16) ? 0 : 8;
  const _Float16* p = base + (size_t)n * stride + koff;
  v8h lo = *(const v8h*)(p);
  v8h hi = *(const v8h*)(p + 16);
  v16h r;
#pragma unroll
  for (int i = 0; i < 8; ++i) { r[i] = lo[i]; r[i + 8] = hi[i]; }
  return r;
}

__device__ inline float gelu_tanh(float v) {
  const float c0 = 0.7978845608028654f; // sqrt(2/pi)
  return 0.5f * v * (1.0f + tanhf(c0 * (v + 0.044715f * v * v * v)));
}

// ------------------------------------------------------ weight transpose ----
__global__ void wtrans_f16(const float* __restrict__ W, _Float16* __restrict__ Wt,
                           int K, int N) {
  size_t idx = (size_t)blockIdx.x * blockDim.x + threadIdx.x;
  size_t tot = (size_t)K * N;
  if (idx >= tot) return;
  int n = (int)(idx / K);
  int k = (int)(idx % K);
  Wt[idx] = (_Float16)W[(size_t)k * N + n];
}

// ----------------------------------------------------------- layernorm ------
__global__ void ln_to_f16(const float* __restrict__ x, const float* __restrict__ g,
                          const float* __restrict__ s, _Float16* __restrict__ out) {
  const int row = blockIdx.x;
  const int tid = threadIdx.x;
  const float* xr = x + (size_t)row * DDIM;
  float sum = 0.f, sq = 0.f;
#pragma unroll
  for (int c = tid; c < DDIM; c += 256) { float v = xr[c]; sum += v; sq += v * v; }
#pragma unroll
  for (int m = 16; m >= 1; m >>= 1) {
    sum += __shfl_xor(sum, m);
    sq  += __shfl_xor(sq, m);
  }
  __shared__ float sA[8], sB[8];
  if ((tid & 31) == 0) { sA[tid >> 5] = sum; sB[tid >> 5] = sq; }
  __syncthreads();
  float ts = 0.f, tq = 0.f;
#pragma unroll
  for (int i = 0; i < 8; ++i) { ts += sA[i]; tq += sB[i]; }
  const float mean = ts * (1.0f / DDIM);
  const float var  = tq * (1.0f / DDIM) - mean * mean;
  const float rstd = rsqrtf(var + 1e-5f);
  _Float16* orow = out + (size_t)row * DDIM;
  for (int c = tid; c < DDIM; c += 256)
    orow[c] = (_Float16)((xr[c] - mean) * rstd * g[c] + s[c]);
}

// ------------------------------------------------------------- GEMM ---------
// C[M,N] = A[M,K](f16) * Bt[N,K](f16)^T (+bias)(+gelu)(+res)
// 128x128x32 block tile, 8 waves (32x64 each), double-buffered LDS fed by
// GLOBAL_LOAD_ASYNC_TO_LDS_B128 (4 async ops / thread / tile, ASYNCcnt).
__device__ inline void gemm_stage_async(const _Float16* A, const _Float16* Bt,
                                        _Float16* As, _Float16* Bs,
                                        int rowBase, int colBase, int K, int kt,
                                        int tid) {
#pragma unroll
  for (int i = 0; i < 2; ++i) {
    int idx = tid + i * 256;
    int r   = idx >> 2;
    int ch  = idx & 3;
    async_ld_b128(&As[r * 40 + ch * 8], A  + (size_t)(rowBase + r) * K + kt + ch * 8);
    async_ld_b128(&Bs[r * 40 + ch * 8], Bt + (size_t)(colBase + r) * K + kt + ch * 8);
  }
}

__global__ __launch_bounds__(256)
void gemm_wmma_f16(const _Float16* __restrict__ A, const _Float16* __restrict__ Bt,
                   const float* __restrict__ bias, const float* __restrict__ res,
                   float* __restrict__ outF, _Float16* __restrict__ outH,
                   int M, int N, int K, int doGelu) {
  __shared__ __align__(16) _Float16 As[2][128 * 40];
  __shared__ __align__(16) _Float16 Bs[2][128 * 40];

  const int tid  = threadIdx.x;
  const int lane = tid & 31;
  const int wave = tid >> 5;
  const int wm   = wave >> 1;
  const int wn   = wave & 1;
  const int rowBase = blockIdx.y * 128;
  const int colBase = blockIdx.x * 128;

  v8f acc[2][4];
#pragma unroll
  for (int mi = 0; mi < 2; ++mi)
#pragma unroll
    for (int ni = 0; ni < 4; ++ni) acc[mi][ni] = vzero8();

  // prologue: async-stage tile 0 into buffer 0
  gemm_stage_async(A, Bt, As[0], Bs[0], rowBase, colBase, K, 0, tid);

  int ib = 0;
  for (int kt = 0; kt < K; kt += 32) {
    const bool hasNext = (kt + 32 < K);
    if (hasNext) {
      gemm_stage_async(A, Bt, As[ib ^ 1], Bs[ib ^ 1], rowBase, colBase, K,
                       kt + 32, tid);
      // keep the 4 in-flight next-tile copies; current tile's 4 are done
      ASYNC_WAIT_KEEP4();
      // warm L2 for tile kt+64 (near-cache hint)
      if (kt + 64 < K) {
        __builtin_prefetch(A  + (size_t)(rowBase + (tid >> 1)) * K + kt + 64, 0, 3);
        __builtin_prefetch(Bt + (size_t)(colBase + (tid >> 1)) * K + kt + 64, 0, 3);
      }
    } else {
      ASYNC_WAIT_ALL();
    }
    __syncthreads();   // current tile visible to all waves

    v16h af[2], bf[4];
#pragma unroll
    for (int mi = 0; mi < 2; ++mi)
      af[mi] = load_frag_a(&As[ib][(wm * 32 + mi * 16) * 40], 40, lane);
#pragma unroll
    for (int ni = 0; ni < 4; ++ni)
      bf[ni] = load_frag_b(&Bs[ib][(wn * 64 + ni * 16) * 40], 40, lane);
#pragma unroll
    for (int mi = 0; mi < 2; ++mi)
#pragma unroll
      for (int ni = 0; ni < 4; ++ni)
        acc[mi][ni] = wmma_f16f32(af[mi], bf[ni], acc[mi][ni]);

    __syncthreads();   // all waves done reading buf[ib] before it is re-staged
    ib ^= 1;
  }

  // epilogue — C layout: lane n=lane&15 ; vgpr r -> m = r + (lane<16?0:8)
  const int nlan = lane & 15;
  const int hi   = (lane < 16) ? 0 : 8;
#pragma unroll
  for (int mi = 0; mi < 2; ++mi) {
#pragma unroll
    for (int ni = 0; ni < 4; ++ni) {
      const int col = colBase + wn * 64 + ni * 16 + nlan;
      const float bv = bias ? bias[col] : 0.0f;
#pragma unroll
      for (int r = 0; r < 8; ++r) {
        const int row = rowBase + wm * 32 + mi * 16 + r + hi;
        float v = acc[mi][ni][r] + bv;
        if (doGelu) v = gelu_tanh(v);
        if (res)  v += res[(size_t)row * N + col];
        const size_t o = (size_t)row * N + col;
        if (outF) outF[o] = v;
        if (outH) outH[o] = (_Float16)v;
      }
    }
  }
}

// -------------------------------------------------- flash attention ---------
// One block = (b, h, 64 query rows); 4 waves; each wave owns 16 q rows.
__global__ __launch_bounds__(128)
void attn_flash(const _Float16* __restrict__ Q, const _Float16* __restrict__ Kf,
                const _Float16* __restrict__ Vf, _Float16* __restrict__ O) {
  const int b  = blockIdx.z, h = blockIdx.y, qb = blockIdx.x;
  const int tid = threadIdx.x, wave = tid >> 5, lane = tid & 31;
  const int nlan = lane & 15;
  const int hi   = (lane < 16) ? 0 : 8;
  const int qrow0 = qb * 64 + wave * 16;

  __shared__ __align__(16) _Float16 Ks[32 * 72];      // [key][d]
  __shared__ __align__(16) _Float16 Vt[64 * 40];      // [d][key] (transposed)
  __shared__ __align__(16) _Float16 Ps[4][16 * 40];   // per-wave P tile [m][key]

  v16h qf[2];
#pragma unroll
  for (int kk = 0; kk < 2; ++kk)
    qf[kk] = load_frag_a(Q + ((size_t)b * TDIM + qrow0) * DDIM + h * HD + kk * 32,
                         DDIM, lane);

  float rmax[8], rsum[8];
  v8f acc[4];
#pragma unroll
  for (int r = 0; r < 8; ++r) { rmax[r] = -1e30f; rsum[r] = 0.0f; }
#pragma unroll
  for (int di = 0; di < 4; ++di) acc[di] = vzero8();

  const int nCh = (qb + 1) * 2;   // uniform trip count (masked tail)
  for (int kc = 0; kc < nCh; ++kc) {
    const int key0 = kc * 32;
    __syncthreads();
    // stage K (async, row-major) and V (sync + transposed ds stores)
#pragma unroll
    for (int i = 0; i < 2; ++i) {
      int idx = tid + i * 128;
      int key = idx >> 3;
      int c   = idx & 7;
      const size_t grow = ((size_t)b * TDIM + key0 + key) * DDIM + h * HD + c * 8;
      async_ld_b128(&Ks[key * 72 + c * 8], Kf + grow);
      v8h vv = *(const v8h*)(Vf + grow);
#pragma unroll
      for (int j = 0; j < 8; ++j) Vt[(c * 8 + j) * 40 + key] = vv[j];
    }
    ASYNC_WAIT_ALL();
    __syncthreads();

    // S = Q K^T : 16 q-rows x 32 keys (2 tiles x 2 k-steps)
    v8f sf[2];
    sf[0] = vzero8(); sf[1] = vzero8();
#pragma unroll
    for (int kk = 0; kk < 2; ++kk)
#pragma unroll
      for (int ni = 0; ni < 2; ++ni)
        sf[ni] = wmma_f16f32(qf[kk],
                             load_frag_b(&Ks[(ni * 16) * 72 + kk * 32], 72, lane),
                             sf[ni]);

    float sv[2][8];
#pragma unroll
    for (int ni = 0; ni < 2; ++ni)
#pragma unroll
      for (int r = 0; r < 8; ++r) {
        float v = sf[ni][r] * 0.125f;           // 1/sqrt(64)
        const int kg = key0 + ni * 16 + nlan;
        const int qg = qrow0 + r + hi;
        sv[ni][r] = (kg > qg) ? -1e30f : v;
      }

    float corr[8];
#pragma unroll
    for (int r = 0; r < 8; ++r) {
      float cm = fmaxf(sv[0][r], sv[1][r]);
#pragma unroll
      for (int m = 8; m >= 1; m >>= 1) cm = fmaxf(cm, __shfl_xor(cm, m));
      const float nm = fmaxf(rmax[r], cm);
      corr[r] = __expf(rmax[r] - nm);
      rmax[r] = nm;
    }
    float pv[2][8], csum[8];
#pragma unroll
    for (int r = 0; r < 8; ++r) csum[r] = 0.0f;
#pragma unroll
    for (int ni = 0; ni < 2; ++ni)
#pragma unroll
      for (int r = 0; r < 8; ++r) {
        float p = __expf(sv[ni][r] - rmax[r]);
        pv[ni][r] = p;
        csum[r] += p;
      }
#pragma unroll
    for (int r = 0; r < 8; ++r) {
#pragma unroll
      for (int m = 8; m >= 1; m >>= 1) csum[r] += __shfl_xor(csum[r], m);
      rsum[r] = rsum[r] * corr[r] + csum[r];
    }
#pragma unroll
    for (int di = 0; di < 4; ++di)
#pragma unroll
      for (int r = 0; r < 8; ++r) acc[di][r] *= corr[r];

    // P (C-layout) -> LDS -> A-fragment for PV
#pragma unroll
    for (int ni = 0; ni < 2; ++ni)
#pragma unroll
      for (int r = 0; r < 8; ++r)
        Ps[wave][(r + hi) * 40 + ni * 16 + nlan] = (_Float16)pv[ni][r];
    __syncthreads();

    v16h pf = load_frag_a(&Ps[wave][0], 40, lane);
#pragma unroll
    for (int di = 0; di < 4; ++di)
      acc[di] = wmma_f16f32(pf, load_frag_b(&Vt[(di * 16) * 40], 40, lane), acc[di]);
  }

#pragma unroll
  for (int di = 0; di < 4; ++di)
#pragma unroll
    for (int r = 0; r < 8; ++r) {
      const float o = acc[di][r] / rsum[r];
      O[((size_t)b * TDIM + qrow0 + r + hi) * DDIM + h * HD + di * 16 + nlan] =
          (_Float16)o;
    }
}

// ----------------------------------------------------------- launcher -------
extern "C" void kernel_launch(void* const* d_in, const int* in_sizes, int n_in,
                              void* d_out, int out_size, void* d_ws, size_t ws_size,
                              hipStream_t stream) {
  (void)in_sizes; (void)n_in; (void)out_size; (void)ws_size;
  const float* x  = (const float*)d_in[0];
  const float* Wq = (const float*)d_in[1];
  const float* Wk = (const float*)d_in[2];
  const float* Wv = (const float*)d_in[3];
  const float* Wo = (const float*)d_in[4];
  const float* bo = (const float*)d_in[5];
  const float* W1 = (const float*)d_in[6];
  const float* b1 = (const float*)d_in[7];
  const float* W2 = (const float*)d_in[8];
  const float* b2 = (const float*)d_in[9];
  const float* g1 = (const float*)d_in[10];
  const float* s1 = (const float*)d_in[11];
  const float* g2 = (const float*)d_in[12];
  const float* s2 = (const float*)d_in[13];
  float* out = (float*)d_out;

  char* ws = (char*)d_ws;
  size_t off = 0;
  auto alloc = [&](size_t bytes) -> void* {
    void* p = ws + off;
    off = (off + bytes + 255) & ~(size_t)255;
    return p;
  };
  _Float16* WqT = (_Float16*)alloc((size_t)DDIM * DDIM * 2);
  _Float16* WkT = (_Float16*)alloc((size_t)DDIM * DDIM * 2);
  _Float16* WvT = (_Float16*)alloc((size_t)DDIM * DDIM * 2);
  _Float16* WoT = (_Float16*)alloc((size_t)DDIM * DDIM * 2);
  _Float16* W1T = (_Float16*)alloc((size_t)DDIM * DFF * 2);
  _Float16* W2T = (_Float16*)alloc((size_t)DFF * DDIM * 2);
  _Float16* h1  = (_Float16*)alloc((size_t)MROWS * DDIM * 2);
  _Float16* Qh  = (_Float16*)alloc((size_t)MROWS * DDIM * 2);
  _Float16* Kh  = (_Float16*)alloc((size_t)MROWS * DDIM * 2);
  _Float16* Vh  = (_Float16*)alloc((size_t)MROWS * DDIM * 2);
  _Float16* Ctx = (_Float16*)alloc((size_t)MROWS * DDIM * 2);
  float*    xr  = (float*)   alloc((size_t)MROWS * DDIM * 4);
  _Float16* h2  = (_Float16*)alloc((size_t)MROWS * DDIM * 2);
  _Float16* ff1 = (_Float16*)alloc((size_t)MROWS * DFF * 2);

  {
    int totD = DDIM * DDIM, totF = DDIM * DFF;
    wtrans_f16<<<(totD + 255) / 256, 256, 0, stream>>>(Wq, WqT, DDIM, DDIM);
    wtrans_f16<<<(totD + 255) / 256, 256, 0, stream>>>(Wk, WkT, DDIM, DDIM);
    wtrans_f16<<<(totD + 255) / 256, 256, 0, stream>>>(Wv, WvT, DDIM, DDIM);
    wtrans_f16<<<(totD + 255) / 256, 256, 0, stream>>>(Wo, WoT, DDIM, DDIM);
    wtrans_f16<<<(totF + 255) / 256, 256, 0, stream>>>(W1, W1T, DDIM, DFF);
    wtrans_f16<<<(totF + 255) / 256, 256, 0, stream>>>(W2, W2T, DFF, DDIM);
  }
  ln_to_f16<<<MROWS, 256, 0, stream>>>(x, g1, s1, h1);
  dim3 gD(DDIM / 128, MROWS / 128);
  gemm_wmma_f16<<<gD, 256, 0, stream>>>(h1, WqT, nullptr, nullptr, nullptr, Qh,
                                        MROWS, DDIM, DDIM, 0);
  gemm_wmma_f16<<<gD, 256, 0, stream>>>(h1, WkT, nullptr, nullptr, nullptr, Kh,
                                        MROWS, DDIM, DDIM, 0);
  gemm_wmma_f16<<<gD, 256, 0, stream>>>(h1, WvT, nullptr, nullptr, nullptr, Vh,
                                        MROWS, DDIM, DDIM, 0);
  attn_flash<<<dim3(TDIM / 64, HDIM, BDIM), 128, 0, stream>>>(Qh, Kh, Vh, Ctx);
  gemm_wmma_f16<<<gD, 256, 0, stream>>>(Ctx, WoT, bo, x, xr, nullptr,
                                        MROWS, DDIM, DDIM, 0);
  ln_to_f16<<<MROWS, 256, 0, stream>>>(xr, g2, s2, h2);
  dim3 gF(DFF / 128, MROWS / 128);
  gemm_wmma_f16<<<gF, 256, 0, stream>>>(h2, W1T, b1, nullptr, nullptr, ff1,
                                        MROWS, DFF, DDIM, 1);
  gemm_wmma_f16<<<gD, 256, 0, stream>>>(ff1, W2T, b2, xr, out, nullptr,
                                        MROWS, DDIM, DFF, 0);
}
